// LinearAttention_18373870092512
// MI455X (gfx1250) — matrix operand order repaired and checked by reading, not compiled, not run on hardware
//
#include <hip/hip_runtime.h>
#include <stdint.h>

#define B_    8
#define C_    512
#define L_    8192
#define H_    8
#define DH_   64
#define OC3_  1536
#define RSQ_C 22.627416998f   /* sqrt(512) */

typedef __attribute__((ext_vector_type(8)))  float        v8f;
typedef __attribute__((ext_vector_type(16))) __bf16       v16bf;
typedef __attribute__((ext_vector_type(4)))  unsigned int u32x4;
typedef __attribute__((ext_vector_type(8)))  int          i32x8;
typedef __attribute__((ext_vector_type(4)))  int          i32x4;

#if defined(__has_builtin)
#if __has_builtin(__builtin_amdgcn_tensor_load_to_lds) && __has_builtin(__builtin_amdgcn_s_wait_tensorcnt)
#define HAS_TDM 1
#endif
#endif
#ifndef HAS_TDM
#define HAS_TDM 0
#endif

#define LDSOFF(p) ((uint32_t)(uintptr_t)(p))

union FragB16 { v16bf v; uint32_t u[8]; uint4 q[2]; };
union FragAcc { v8f   v; float    f[8]; };

__device__ __forceinline__ uint16_t f2bf(float f) {
  uint32_t u = __float_as_uint(f);
  u += 0x7FFFu + ((u >> 16) & 1u);          // round-to-nearest-even
  return (uint16_t)(u >> 16);
}
__device__ __forceinline__ float bf2f(uint16_t h) {
  return __uint_as_float(((uint32_t)h) << 16);
}
__device__ __forceinline__ uint32_t pack2(float lo, float hi) {
  return (uint32_t)f2bf(lo) | ((uint32_t)f2bf(hi) << 16);
}
__device__ __forceinline__ v8f wmma_bf16(const FragB16& a, const FragB16& b, v8f c) {
  return __builtin_amdgcn_wmma_f32_16x16x32_bf16(false, a.v, false, b.v,
                                                 (short)0, c, false, false);
}

/* ---- TDM: 2-D tile (data_size=4B) global->LDS with row padding -------- */
/* pad_int code 3 = pad after every 16 dwords; pad_amt = (dwords-1)        */
__device__ __forceinline__ void tdm_load_2d(uint32_t lds_off, uint64_t gaddr,
                                            uint32_t tile_d0, uint32_t tile_d1,
                                            uint32_t tdim0, uint32_t tdim1,
                                            uint32_t stride0,
                                            uint32_t pad_int, uint32_t pad_amt) {
#if HAS_TDM
  u32x4 g0;
  g0[0] = 1u;                                              // count=1
  g0[1] = lds_off;                                         // lds_addr
  g0[2] = (uint32_t)gaddr;                                 // global_addr lo
  g0[3] = ((uint32_t)(gaddr >> 32) & 0x01FFFFFFu) | (2u << 30); // hi | type=2
  i32x8 g1;
  g1[0] = (int)((2u << 16) | (1u << 20) | (pad_int << 22) | (pad_amt << 25));
  g1[1] = (int)(tdim0 << 16);                              // tensor_dim0 lo16
  g1[2] = (int)((tdim0 >> 16) | (tdim1 << 16));            // dim0 hi / dim1 lo
  g1[3] = (int)((tdim1 >> 16) | (tile_d0 << 16));          // dim1 hi / tile_d0
  g1[4] = (int)(tile_d1 & 0xFFFFu);                        // tile_d1, tile_d2=0
  g1[5] = (int)stride0;                                    // dim0_stride lo32
  g1[6] = 0;
  g1[7] = 0;
  i32x4 z4 = {0, 0, 0, 0};
  i32x8 z8 = {0, 0, 0, 0, 0, 0, 0, 0};
  __builtin_amdgcn_tensor_load_to_lds(g0, g1, z4, z4, z8, 0);
#else
  (void)lds_off; (void)gaddr; (void)tile_d0; (void)tile_d1;
  (void)tdim0; (void)tdim1; (void)stride0; (void)pad_int; (void)pad_amt;
#endif
}

/* ---------------- K0: pre-pack W matrices to bf16 pairs ----------------- */
__global__ __launch_bounds__(256) void k_pack_w(const float* __restrict__ Wq,
                                                const float* __restrict__ Wo,
                                                uint32_t* __restrict__ wq_p,
                                                uint32_t* __restrict__ wo_p) {
  int t = blockIdx.x * 256 + threadIdx.x;          // (1536+512)*256
  if (t < OC3_ * 256) {
    wq_p[t] = pack2(Wq[2 * t], Wq[2 * t + 1]);
  } else {
    int u = t - OC3_ * 256;
    wo_p[u] = pack2(Wo[2 * u], Wo[2 * u + 1]);
  }
}

/* ---------------- K1a: channel L2 norms of x ---------------- */
__global__ void k_norm1(const float* __restrict__ x, float* __restrict__ rnorm) {
  int t = blockIdx.x * blockDim.x + threadIdx.x;   // B_*L_
  int b = t / L_, l = t % L_;
  const float* p = x + (size_t)b * C_ * L_ + l;
  float s = 0.f;
#pragma unroll 4
  for (int c = 0; c < C_; ++c) { float v = p[(size_t)c * L_]; s += v * v; }
  rnorm[t] = RSQ_C / fmaxf(sqrtf(s), 1e-12f);
}

/* -------- K1b: scale + bf16 pack into blocked layout [b][kb][l][16] ---- */
__global__ __launch_bounds__(256) void k_scale_pack_x(const float* __restrict__ x,
                                                      const float* __restrict__ g1,
                                                      const float* __restrict__ rnorm,
                                                      uint32_t* __restrict__ xn_p) {
  int l  = blockIdx.x * 256 + threadIdx.x;
  int kb = blockIdx.y;                                   // 0..15
  int b  = blockIdx.z;
  float rn = rnorm[b * L_ + l];
  uint32_t o[16];
#pragma unroll
  for (int i = 0; i < 16; ++i) {
    int c = kb * 32 + 2 * i;
    float v0 = x[((size_t)b * C_ + c) * L_ + l]     * rn * g1[c];
    float v1 = x[((size_t)b * C_ + c + 1) * L_ + l] * rn * g1[c + 1];
    o[i] = pack2(v0, v1);
  }
  uint4* dst = (uint4*)(xn_p + (((size_t)(b * 16 + kb)) * L_ + l) * 16);
#pragma unroll
  for (int q = 0; q < 4; ++q)
    dst[q] = make_uint4(o[4 * q], o[4 * q + 1], o[4 * q + 2], o[4 * q + 3]);
}

/* ---------------- K2: qkv = Wqkv(1536x512) * xn(512xL), bf16 WMMA ------- */
__global__ __launch_bounds__(256) void k_gemm_qkv(const uint32_t* __restrict__ w_p,
                                                  const uint32_t* __restrict__ xn_p,
                                                  uint16_t* __restrict__ qkv) {
  alignas(16) __shared__ uint32_t ldsA[2][128][20];       // [m][k2] + pad
  alignas(16) __shared__ uint32_t ldsB[2][128][20];       // [n][k2] + pad
  int b  = blockIdx.z;
  int m0 = blockIdx.y * 128;
  int n0 = blockIdx.x * 128;
  int t = threadIdx.x;
  int wave = t >> 5, lane = t & 31;
  int half = lane >> 4, lidx = lane & 15;

  FragAcc acc[8];
#pragma unroll
  for (int i = 0; i < 8; ++i)
#pragma unroll
    for (int r = 0; r < 8; ++r) acc[i].f[r] = 0.f;

#if HAS_TDM
  if (t < 32) {
    tdm_load_2d(LDSOFF(&ldsA[0][0][0]),
                (uint64_t)(uintptr_t)(w_p + (size_t)m0 * 256),
                16, 128, 256, 128, 256, 3, 3);
    tdm_load_2d(LDSOFF(&ldsB[0][0][0]),
                (uint64_t)(uintptr_t)(xn_p + ((size_t)(b * 16) * L_ + n0) * 16),
                16, 128, 16, 128, 16, 3, 3);
  }
#endif
  for (int kb = 0; kb < 16; ++kb) {
    int buf = kb & 1;
#if HAS_TDM
    if (t < 32) __builtin_amdgcn_s_wait_tensorcnt(0);
#else
#pragma unroll
    for (int i = 0; i < 8; ++i) {                         // manual A copy
      int idx = t * 8 + i;
      int m = idx >> 4, k2 = idx & 15;
      ldsA[buf][m][k2] = w_p[(size_t)(m0 + m) * 256 + kb * 16 + k2];
    }
#pragma unroll
    for (int i = 0; i < 8; ++i) {                         // manual B copy
      int idx = i * 256 + t;
      int n = idx >> 4, j = idx & 15;
      ldsB[buf][n][j] = xn_p[((size_t)(b * 16 + kb) * L_ + n0) * 16 + idx];
    }
#endif
    __syncthreads();
#if HAS_TDM
    if (kb + 1 < 16 && t < 32) {
      tdm_load_2d(LDSOFF(&ldsA[buf ^ 1][0][0]),
                  (uint64_t)(uintptr_t)(w_p + (size_t)m0 * 256 + (kb + 1) * 16),
                  16, 128, 256, 128, 256, 3, 3);
      tdm_load_2d(LDSOFF(&ldsB[buf ^ 1][0][0]),
                  (uint64_t)(uintptr_t)(xn_p + ((size_t)(b * 16 + kb + 1) * L_ + n0) * 16),
                  16, 128, 16, 128, 16, 3, 3);
    }
#endif
    FragB16 a;
    a.q[0] = *(const uint4*)&ldsA[buf][wave * 16 + lidx][half * 4];
    a.q[1] = *(const uint4*)&ldsA[buf][wave * 16 + lidx][8 + half * 4];
#pragma unroll
    for (int nt = 0; nt < 8; ++nt) {
      FragB16 bf;
      int n = nt * 16 + lidx;
      bf.q[0] = *(const uint4*)&ldsB[buf][n][half * 4];
      bf.q[1] = *(const uint4*)&ldsB[buf][n][8 + half * 4];
      acc[nt].v = wmma_bf16(a, bf, acc[nt].v);
    }
    __syncthreads();
  }
  if (blockIdx.y < 4) {                        // q region: blocked pair-packed
    uint32_t* qblk = (uint32_t*)qkv + (size_t)b * 768 * L_;
    int rp = (m0 + wave * 16 + half * 8) >> 1; // multiple of 4
    int kbq = rp >> 4, jb = rp & 15;
#pragma unroll
    for (int nt = 0; nt < 8; ++nt) {
      int col = n0 + nt * 16 + lidx;
      uint4 o = make_uint4(pack2(acc[nt].f[0], acc[nt].f[1]),
                           pack2(acc[nt].f[2], acc[nt].f[3]),
                           pack2(acc[nt].f[4], acc[nt].f[5]),
                           pack2(acc[nt].f[6], acc[nt].f[7]));
      *(uint4*)&qblk[((size_t)kbq * L_ + col) * 16 + jb] = o;
    }
  } else {                                     // k,v: plain bf16 rows
#pragma unroll
    for (int nt = 0; nt < 8; ++nt) {
      int col = n0 + nt * 16 + lidx;
#pragma unroll
      for (int r = 0; r < 8; ++r) {
        int row = m0 + wave * 16 + half * 8 + r;
        qkv[((size_t)b * OC3_ + row) * L_ + col] = f2bf(acc[nt].f[r]);
      }
    }
  }
}

/* ------- K3q: softmax over head-dim (blocked q layout, in place) -------- */
__global__ void k_softmax_q(uint16_t* qkv) {
  int t = blockIdx.x * blockDim.x + threadIdx.x;   // B_*H_*L_
  int l  = t % L_;
  int bh = t / L_;
  int h = bh % H_, b = bh / H_;
  uint32_t* qblk = (uint32_t*)qkv + (size_t)b * 768 * L_;
  size_t base0 = ((size_t)(2 * h)     * L_ + l) * 16;
  size_t base1 = ((size_t)(2 * h + 1) * L_ + l) * 16;
  uint32_t v[32];
#pragma unroll
  for (int i = 0; i < 4; ++i) *(uint4*)&v[i * 4]      = *(const uint4*)&qblk[base0 + i * 4];
#pragma unroll
  for (int i = 0; i < 4; ++i) *(uint4*)&v[16 + i * 4] = *(const uint4*)&qblk[base1 + i * 4];
  float mx = -1e30f;
#pragma unroll
  for (int j = 0; j < 32; ++j) {
    mx = fmaxf(mx, bf2f((uint16_t)(v[j] & 0xFFFFu)));
    mx = fmaxf(mx, bf2f((uint16_t)(v[j] >> 16)));
  }
  float s = 0.f;
#pragma unroll
  for (int j = 0; j < 32; ++j) {
    s += __expf(bf2f((uint16_t)(v[j] & 0xFFFFu)) - mx);
    s += __expf(bf2f((uint16_t)(v[j] >> 16)) - mx);
  }
  float inv = 0.125f / s;                          // * dh^-0.5
#pragma unroll
  for (int j = 0; j < 32; ++j) {
    float a = __expf(bf2f((uint16_t)(v[j] & 0xFFFFu)) - mx) * inv;
    float c = __expf(bf2f((uint16_t)(v[j] >> 16)) - mx) * inv;
    v[j] = pack2(a, c);
  }
#pragma unroll
  for (int i = 0; i < 4; ++i) *(uint4*)&qblk[base0 + i * 4] = *(const uint4*)&v[i * 4];
#pragma unroll
  for (int i = 0; i < 4; ++i) *(uint4*)&qblk[base1 + i * 4] = *(const uint4*)&v[16 + i * 4];
}

/* ---------------- K3k: softmax over sequence (one block per row) -------- */
__global__ __launch_bounds__(256) void k_softmax_k(uint16_t* qkv) {
  __shared__ float red[256];
  int row = blockIdx.x;                            // B_*H_*DH_
  int d = row % DH_, h = (row / DH_) % H_, b = row / (DH_ * H_);
  uint16_t* p = qkv + ((size_t)b * OC3_ + 512 + h * 64 + d) * L_;
  int t = threadIdx.x;
  float vals[32];
  float mx = -1e30f;
#pragma unroll
  for (int i = 0; i < 32; ++i) { vals[i] = bf2f(p[t + i * 256]); mx = fmaxf(mx, vals[i]); }
  red[t] = mx; __syncthreads();
  for (int s = 128; s > 0; s >>= 1) { if (t < s) red[t] = fmaxf(red[t], red[t + s]); __syncthreads(); }
  mx = red[0]; __syncthreads();
  float sum = 0.f;
#pragma unroll
  for (int i = 0; i < 32; ++i) { vals[i] = __expf(vals[i] - mx); sum += vals[i]; }
  red[t] = sum; __syncthreads();
  for (int s = 128; s > 0; s >>= 1) { if (t < s) red[t] += red[t + s]; __syncthreads(); }
  float inv = 1.f / red[0];
#pragma unroll
  for (int i = 0; i < 32; ++i) p[t + i * 256] = f2bf(vals[i] * inv);
}

/* ---------------- K4: context^T[e][d] = sum_n k[d,n] v[e,n] ------------- */
__global__ __launch_bounds__(128) void k_context(const uint16_t* __restrict__ qkv,
                                                 uint16_t* __restrict__ ctxT) {
  alignas(16) __shared__ uint32_t ldsA[2][64][20];        // k rows [d][npairs]
  alignas(16) __shared__ uint32_t ldsB[2][64][20];        // v rows [e][npairs]
  int bh = blockIdx.x;
  int h = bh % H_, b = bh / H_;
  int t = threadIdx.x;
  int wave = t >> 5, lane = t & 31;
  int half = lane >> 4, lidx = lane & 15;
  const uint16_t* kb16 = qkv + ((size_t)b * OC3_ +  512 + h * 64) * L_;
  const uint16_t* vb16 = qkv + ((size_t)b * OC3_ + 1024 + h * 64) * L_;
  FragAcc acc[4];
#pragma unroll
  for (int i = 0; i < 4; ++i)
#pragma unroll
    for (int r = 0; r < 8; ++r) acc[i].f[r] = 0.f;

#if HAS_TDM
  if (t < 32) {
    tdm_load_2d(LDSOFF(&ldsA[0][0][0]), (uint64_t)(uintptr_t)kb16,
                16, 64, L_ / 2, 64, L_ / 2, 3, 3);
    tdm_load_2d(LDSOFF(&ldsB[0][0][0]), (uint64_t)(uintptr_t)vb16,
                16, 64, L_ / 2, 64, L_ / 2, 3, 3);
  }
#endif
  for (int it = 0; it < L_ / 32; ++it) {
    int buf = it & 1;
#if HAS_TDM
    if (t < 32) __builtin_amdgcn_s_wait_tensorcnt(0);
#else
    int n0 = it * 32;
#pragma unroll
    for (int i = 0; i < 8; ++i) {
      int idx = i * 128 + t;
      int d = idx >> 4, k2 = idx & 15;
      ldsA[buf][d][k2] = *(const uint32_t*)(kb16 + (size_t)d * L_ + n0 + 2 * k2);
      ldsB[buf][d][k2] = *(const uint32_t*)(vb16 + (size_t)d * L_ + n0 + 2 * k2);
    }
#endif
    __syncthreads();
#if HAS_TDM
    if (it + 1 < L_ / 32 && t < 32) {
      uint64_t off = (uint64_t)(it + 1) * 64;             // bytes: 32 elems*2B
      tdm_load_2d(LDSOFF(&ldsA[buf ^ 1][0][0]), (uint64_t)(uintptr_t)kb16 + off,
                  16, 64, L_ / 2, 64, L_ / 2, 3, 3);
      tdm_load_2d(LDSOFF(&ldsB[buf ^ 1][0][0]), (uint64_t)(uintptr_t)vb16 + off,
                  16, 64, L_ / 2, 64, L_ / 2, 3, 3);
    }
#endif
    FragB16 a;
    a.q[0] = *(const uint4*)&ldsA[buf][wave * 16 + lidx][half * 4];
    a.q[1] = *(const uint4*)&ldsA[buf][wave * 16 + lidx][8 + half * 4];
#pragma unroll
    for (int et = 0; et < 4; ++et) {
      FragB16 bf;
      bf.q[0] = *(const uint4*)&ldsB[buf][et * 16 + lidx][half * 4];
      bf.q[1] = *(const uint4*)&ldsB[buf][et * 16 + lidx][8 + half * 4];
      acc[et].v = wmma_bf16(a, bf, acc[et].v);
    }
    __syncthreads();
  }
#pragma unroll
  for (int et = 0; et < 4; ++et)
#pragma unroll
    for (int r = 0; r < 8; ++r) {
      int e = et * 16 + lidx;
      int d = wave * 16 + half * 8 + r;
      ctxT[(size_t)bh * 4096 + e * 64 + d] = f2bf(acc[et].f[r]);
    }
}

/* ---------------- K5: out[e,n] = sum_d ctxT[e,d] q[d,n] ----------------- */
__global__ __launch_bounds__(256) void k_attn_out(const uint16_t* __restrict__ ctxT,
                                                  const uint16_t* __restrict__ qkv,
                                                  uint32_t* __restrict__ attn_p) {
  alignas(16) __shared__ uint32_t ldsB[256][36];          // [n][32 k2 + pad]
  int bh = blockIdx.y;
  int h = bh % H_, b = bh / H_;
  int n0 = blockIdx.x * 256;
  int t = threadIdx.x;
  int wave = t >> 5, lane = t & 31;
  int half = lane >> 4, lidx = lane & 15;
  const uint32_t* qblk = (const uint32_t*)qkv + (size_t)b * 768 * L_;
#if HAS_TDM
  if (t < 32) {
    tdm_load_2d(LDSOFF(&ldsB[0][0]),
                (uint64_t)(uintptr_t)(qblk + ((size_t)(2 * h) * L_ + n0) * 16),
                16, 256, 16, 256, 16, 3, 19);             // pad 20 dw -> stride 36
    tdm_load_2d(LDSOFF(&ldsB[0][16]),
                (uint64_t)(uintptr_t)(qblk + ((size_t)(2 * h + 1) * L_ + n0) * 16),
                16, 256, 16, 256, 16, 3, 19);
    __builtin_amdgcn_s_wait_tensorcnt(0);
  }
#else
#pragma unroll
  for (int kb2 = 0; kb2 < 2; ++kb2)
#pragma unroll 4
    for (int i = 0; i < 16; ++i) {
      int idx = i * 256 + t;
      int n = idx >> 4, j = idx & 15;
      ldsB[n][kb2 * 16 + j] = qblk[((size_t)(2 * h + kb2) * L_ + n0) * 16 + idx];
    }
#endif
  __syncthreads();
  int mt = wave & 3, nc0 = (wave >> 2) * 128;
  const uint16_t* ab = ctxT + (size_t)bh * 4096 + (mt * 16) * 64;
  FragAcc acc[8];
#pragma unroll
  for (int i = 0; i < 8; ++i)
#pragma unroll
    for (int r = 0; r < 8; ++r) acc[i].f[r] = 0.f;
#pragma unroll
  for (int k0 = 0; k0 < 64; k0 += 32) {
    FragB16 a;
    a.q[0] = *(const uint4*)(ab + (size_t)lidx * 64 + k0 + half * 8);
    a.q[1] = *(const uint4*)(ab + (size_t)lidx * 64 + k0 + 16 + half * 8);
#pragma unroll
    for (int nt = 0; nt < 8; ++nt) {
      FragB16 bf;
      int n = nc0 + nt * 16 + lidx;
      bf.q[0] = *(const uint4*)&ldsB[n][(k0 >> 1) + half * 4];
      bf.q[1] = *(const uint4*)&ldsB[n][(k0 >> 1) + 8 + half * 4];
      acc[nt].v = wmma_bf16(a, bf, acc[nt].v);
    }
  }
  int kbo = 2 * h + (mt >> 1);                  // output kblock
  int jb  = (mt & 1) * 8 + half * 4;            // pair index within block
#pragma unroll
  for (int nt = 0; nt < 8; ++nt) {
    int col = n0 + nc0 + nt * 16 + lidx;
    uint4 o = make_uint4(pack2(acc[nt].f[0], acc[nt].f[1]),
                         pack2(acc[nt].f[2], acc[nt].f[3]),
                         pack2(acc[nt].f[4], acc[nt].f[5]),
                         pack2(acc[nt].f[6], acc[nt].f[7]));
    *(uint4*)&attn_p[((size_t)(b * 16 + kbo) * L_ + col) * 16 + jb] = o;
  }
}

/* ---------------- K6a: y = Wout(512x512)*attn + b_out ------------------- */
__global__ __launch_bounds__(256) void k_gemm_out(const uint32_t* __restrict__ w_p,
                                                  const float* __restrict__ bias,
                                                  const uint32_t* __restrict__ attn_p,
                                                  float* __restrict__ y) {
  alignas(16) __shared__ uint32_t ldsA[2][128][20];
  alignas(16) __shared__ uint32_t ldsB[2][128][20];
  int b  = blockIdx.z;
  int m0 = blockIdx.y * 128;
  int n0 = blockIdx.x * 128;
  int t = threadIdx.x;
  int wave = t >> 5, lane = t & 31;
  int half = lane >> 4, lidx = lane & 15;
  FragAcc acc[8];
#pragma unroll
  for (int i = 0; i < 8; ++i)
#pragma unroll
    for (int r = 0; r < 8; ++r) acc[i].f[r] = 0.f;

#if HAS_TDM
  if (t < 32) {
    tdm_load_2d(LDSOFF(&ldsA[0][0][0]),
                (uint64_t)(uintptr_t)(w_p + (size_t)m0 * 256),
                16, 128, 256, 128, 256, 3, 3);
    tdm_load_2d(LDSOFF(&ldsB[0][0][0]),
                (uint64_t)(uintptr_t)(attn_p + ((size_t)(b * 16) * L_ + n0) * 16),
                16, 128, 16, 128, 16, 3, 3);
  }
#endif
  for (int kb = 0; kb < 16; ++kb) {
    int buf = kb & 1;
#if HAS_TDM
    if (t < 32) __builtin_amdgcn_s_wait_tensorcnt(0);
#else
#pragma unroll
    for (int i = 0; i < 8; ++i) {
      int idx = t * 8 + i;
      int m = idx >> 4, k2 = idx & 15;
      ldsA[buf][m][k2] = w_p[(size_t)(m0 + m) * 256 + kb * 16 + k2];
    }
#pragma unroll
    for (int i = 0; i < 8; ++i) {
      int idx = i * 256 + t;
      int n = idx >> 4, j = idx & 15;
      ldsB[buf][n][j] = attn_p[((size_t)(b * 16 + kb) * L_ + n0) * 16 + idx];
    }
#endif
    __syncthreads();
#if HAS_TDM
    if (kb + 1 < 16 && t < 32) {
      tdm_load_2d(LDSOFF(&ldsA[buf ^ 1][0][0]),
                  (uint64_t)(uintptr_t)(w_p + (size_t)m0 * 256 + (kb + 1) * 16),
                  16, 128, 256, 128, 256, 3, 3);
      tdm_load_2d(LDSOFF(&ldsB[buf ^ 1][0][0]),
                  (uint64_t)(uintptr_t)(attn_p + ((size_t)(b * 16 + kb + 1) * L_ + n0) * 16),
                  16, 128, 16, 128, 16, 3, 3);
    }
#endif
    FragB16 a;
    a.q[0] = *(const uint4*)&ldsA[buf][wave * 16 + lidx][half * 4];
    a.q[1] = *(const uint4*)&ldsA[buf][wave * 16 + lidx][8 + half * 4];
#pragma unroll
    for (int nt = 0; nt < 8; ++nt) {
      FragB16 bf;
      int n = nt * 16 + lidx;
      bf.q[0] = *(const uint4*)&ldsB[buf][n][half * 4];
      bf.q[1] = *(const uint4*)&ldsB[buf][n][8 + half * 4];
      acc[nt].v = wmma_bf16(a, bf, acc[nt].v);
    }
    __syncthreads();
  }
#pragma unroll
  for (int nt = 0; nt < 8; ++nt) {
    int col = n0 + nt * 16 + lidx;
#pragma unroll
    for (int r = 0; r < 8; ++r) {
      int row = m0 + wave * 16 + half * 8 + r;
      y[((size_t)b * C_ + row) * L_ + col] = acc[nt].f[r] + bias[row];
    }
  }
}

/* ---------------- K6b: final RMS norm -> d_out -------------------------- */
__global__ void k_norm2(const float* __restrict__ y, const float* __restrict__ g2,
                        float* __restrict__ out) {
  int t = blockIdx.x * blockDim.x + threadIdx.x;   // B_*L_
  int b = t / L_, l = t % L_;
  const float* p = y + (size_t)b * C_ * L_ + l;
  float s = 0.f;
#pragma unroll 4
  for (int c = 0; c < C_; ++c) { float v = p[(size_t)c * L_]; s += v * v; }
  float r = RSQ_C / fmaxf(sqrtf(s), 1e-12f);
  float* o = out + (size_t)b * C_ * L_ + l;
#pragma unroll 4
  for (int c = 0; c < C_; ++c)
    __builtin_nontemporal_store(p[(size_t)c * L_] * g2[c] * r, &o[(size_t)c * L_]);
}

extern "C" void kernel_launch(void* const* d_in, const int* in_sizes, int n_in,
                              void* d_out, int out_size, void* d_ws, size_t ws_size,
                              hipStream_t stream) {
  const float* x    = (const float*)d_in[0];
  const float* g1   = (const float*)d_in[1];
  const float* wqkv = (const float*)d_in[2];
  const float* wout = (const float*)d_in[3];
  const float* bout = (const float*)d_in[4];
  const float* g2   = (const float*)d_in[5];
  float* out = (float*)d_out;

  char* ws = (char*)d_ws;
  uint32_t* xn_p  = (uint32_t*)ws;                                   // 64 MB (reused as attn_p)
  uint16_t* qkv   = (uint16_t*)(ws + ((size_t)64 << 20));            // 192 MB (y overlays later)
  uint16_t* ctxT  = (uint16_t*)(ws + ((size_t)256 << 20));           // 512 KB
  float*    rnorm = (float*)(ws + ((size_t)256 << 20) + 524288);     // 256 KB
  uint32_t* wq_p  = (uint32_t*)(ws + ((size_t)256 << 20) + 786432);  // 1.5 MB
  uint32_t* wo_p  = (uint32_t*)(ws + ((size_t)256 << 20) + 2359296); // 0.5 MB
  float*    y     = (float*)qkv;

  k_pack_w<<<(OC3_ + C_) * 256 / 256, 256, 0, stream>>>(wqkv, wout, wq_p, wo_p);
  k_norm1<<<(B_ * L_) / 256, 256, 0, stream>>>(x, rnorm);
  k_scale_pack_x<<<dim3(L_ / 256, 16, B_), 256, 0, stream>>>(x, g1, rnorm, xn_p);
  k_gemm_qkv<<<dim3(L_ / 128, OC3_ / 128, B_), 256, 0, stream>>>(wq_p, xn_p, qkv);
  k_softmax_q<<<(B_ * H_ * L_) / 256, 256, 0, stream>>>(qkv);
  k_softmax_k<<<B_ * H_ * DH_, 256, 0, stream>>>(qkv);
  k_context<<<B_ * H_, 128, 0, stream>>>(qkv, ctxT);
  k_attn_out<<<dim3(L_ / 256, B_ * H_), 256, 0, stream>>>(ctxT, qkv, xn_p);
  k_gemm_out<<<dim3(L_ / 128, C_ / 128, B_), 256, 0, stream>>>(wo_p, bout, xn_p, y);
  k_norm2<<<(B_ * L_) / 256, 256, 0, stream>>>(y, g2, out);
}
